// HeteroTCR_24086176596376
// MI455X (gfx1250) — matrix operand order repaired and checked by reading, not compiled, not run on hardware
//
#include <hip/hip_runtime.h>
#include <hip/hip_bf16.h>

// ---------------------------------------------------------------------------
// HeteroTCR forward: 3x bipartite SAGEConv (mean agg) + 3-layer MLP + sigmoid
// bf16-resident pipeline:
//   - one-time: convert inputs + pre-transpose all weights to bf16 [N][K]
//   - GEMMs: v_wmma_f32_16x16x32_bf16, tiles staged via
//     global_load_async_to_lds_b128 with ASYNCcnt double-buffering
//     (issue tile t+1, s_wait_asynccnt 4 -> tile t landed, compute overlaps DMA)
//   - mean aggregation: f32 atomics, then scale+convert to bf16
// ---------------------------------------------------------------------------

typedef __attribute__((ext_vector_type(16))) __bf16 v16bf;
typedef __attribute__((ext_vector_type(8)))  __bf16 v8bf;
typedef __attribute__((ext_vector_type(4)))  __bf16 v4bf;
typedef __attribute__((ext_vector_type(8)))  float  v8f;

#define N_CDR_C 50000
#define N_PEP_C 1000
#define FEAT_C  128
#define HID_C   256
#define MLP_CHUNK 20480   // multiple of 128

// ---------------- one-time conversion kernels ----------------

__global__ void ht_cvt_kernel(const float* __restrict__ in, __bf16* __restrict__ out,
                              long long n) {
    long long i = (long long)blockIdx.x * blockDim.x + threadIdx.x;
    if (i < n) out[i] = (__bf16)in[i];
}

// W [K x N] f32 row-major  ->  Wt [N x K] bf16 row-major
__global__ void ht_wtrans_kernel(const float* __restrict__ W, __bf16* __restrict__ Wt,
                                 int K, int nshift, long long total) {
    long long tid = (long long)blockIdx.x * blockDim.x + threadIdx.x;
    if (tid >= total) return;
    const int n = (int)(tid & ((1 << nshift) - 1));
    const int k = (int)(tid >> nshift);
    Wt[(long long)n * K + k] = (__bf16)W[tid];
}

// ---------------- graph kernels ----------------

__global__ void ht_count_kernel(const int* __restrict__ dst, int E, float* __restrict__ cnt) {
    int i = blockIdx.x * blockDim.x + threadIdx.x;
    if (i < E) atomicAdd(&cnt[dst[i]], 1.0f);
}

__global__ void ht_inv_kernel(float* __restrict__ cnt, int n) {
    int i = blockIdx.x * blockDim.x + threadIdx.x;
    if (i < n) cnt[i] = 1.0f / fmaxf(cnt[i], 1.0f);
}

// one thread per (edge, 4 consecutive features); bf16 src, f32 atomics
__global__ void ht_scatter_kernel(const int* __restrict__ src, const int* __restrict__ dstv,
                                  int E, int d4shift, const __bf16* __restrict__ X,
                                  float* __restrict__ acc, int D) {
    long long tid = (long long)blockIdx.x * blockDim.x + threadIdx.x;
    long long total = (long long)E << d4shift;
    if (tid >= total) return;
    int e = (int)(tid >> d4shift);
    int f = (int)(tid & ((1u << d4shift) - 1u)) * 4;
    const v4bf v = *(const v4bf*)(X + (long long)src[e] * D + f);
    float* a = acc + (long long)dstv[e] * D + f;
    atomicAdd(a + 0, (float)v[0]); atomicAdd(a + 1, (float)v[1]);
    atomicAdd(a + 2, (float)v[2]); atomicAdd(a + 3, (float)v[3]);
}

// agg_bf16 = (acc_f32 * inv_deg[row]) downcast
__global__ void ht_scale_cvt_kernel(const float* __restrict__ acc, const float* __restrict__ inv,
                                    __bf16* __restrict__ out, int n, int dshift) {
    long long tid = (long long)blockIdx.x * blockDim.x + threadIdx.x;
    long long total = (long long)n << dshift;
    if (tid < total) out[tid] = (__bf16)(acc[tid] * inv[tid >> dshift]);
}

// ---------------- WMMA GEMM ----------------
// out[M,N](bf16) = act( A1[idx1][.,K1] @ Wt1^T + A2[idx2][.,K2] @ Wt2^T + bias )
// A bf16 row-major (row stride = K, optional row gather); Wt bf16 [N][ldb].
// Tile 128x128x32, 256 threads (8 waves), each wave -> 64x32 (4x2 WMMA accums).
// LDS double-buffered; tiles staged with global_load_async_to_lds_b128 and
// pipelined on ASYNCcnt (4 async ops in flight per wave per tile).

#define BM 128
#define BN 128
#define BK 32
#define LDSP 40   // padded LDS row stride in halves (bank-conflict free, 16B rows)

template<int ACT>  // 0=none, 1=leaky_relu(0.01), 2=relu
__global__ __launch_bounds__(256)
void ht_gemm2_kernel(const __bf16* __restrict__ A1, const int* __restrict__ idx1, int K1,
                     const __bf16* __restrict__ Wt1, int ldb1,
                     const __bf16* __restrict__ A2, const int* __restrict__ idx2, int K2,
                     const __bf16* __restrict__ Wt2, int ldb2,
                     const float* __restrict__ bias,
                     __bf16* __restrict__ C, int M, int N)
{
    __shared__ __align__(16) __bf16 As[2][BM * LDSP];
    __shared__ __align__(16) __bf16 Bs[2][BN * LDSP];

    const int tid  = threadIdx.x;
    const int lane = tid & 31;
    const int wave = tid >> 5;
    const int wr   = wave >> 2;      // 0..1
    const int wc   = wave & 3;       // 0..3
    const int half = lane >> 4;
    const int l15  = lane & 15;

    const int m_base = blockIdx.y * BM;
    const int n_base = blockIdx.x * BN;

    const int steps1 = K1 / BK;
    const int steps2 = A2 ? (K2 / BK) : 0;
    const int total  = steps1 + steps2;

    // Per-thread staging coords: thread stages one A-row and one Wt-row,
    // two 16B segments each (segments (tid&1)*2 + {0,1} of 4 per 64B row).
    const int segr = tid >> 1;                 // 0..127 : tile row (A) / tile n (B)
    int grow = m_base + segr;
    grow = grow < M ? grow : (M - 1);          // branch-free clamp (C rows >= M never stored)
    const long long arow1 = idx1 ? (long long)idx1[grow] : (long long)grow;
    const long long arow2 = A2 ? (idx2 ? (long long)idx2[grow] : (long long)grow) : 0;

    auto issue_tile = [&](int t, int p) {
        const bool second   = (t >= steps1);
        const __bf16* A     = second ? A2 : A1;
        const __bf16* Wt    = second ? Wt2 : Wt1;
        const int K         = second ? K2 : K1;
        const int ldb       = second ? ldb2 : ldb1;
        const long long ar  = second ? arow2 : arow1;
        const int kk        = (second ? (t - steps1) : t) * BK;
        #pragma unroll
        for (int i = 0; i < 2; ++i) {
            const int ss = (tid & 1) * 2 + i;          // 0..3
            const __bf16* ga = A + ar * K + kk + ss * 8;
            const unsigned la =
                (unsigned)(uintptr_t)(&As[p][segr * LDSP]) + (unsigned)(ss * 16);
            asm volatile("global_load_async_to_lds_b128 %0, %1, off"
                         :: "v"(la), "v"(ga) : "memory");
            const __bf16* gb = Wt + (long long)(n_base + segr) * ldb + kk + ss * 8;
            const unsigned lb =
                (unsigned)(uintptr_t)(&Bs[p][segr * LDSP]) + (unsigned)(ss * 16);
            asm volatile("global_load_async_to_lds_b128 %0, %1, off"
                         :: "v"(lb), "v"(gb) : "memory");
        }
    };

    v8f acc[4][2] = {};

    issue_tile(0, 0);
    for (int t = 0; t < total; ++t) {
        const int p = t & 1;
        if (t + 1 < total) {
            issue_tile(t + 1, p ^ 1);                       // prefetch next tile
            asm volatile("s_wait_asynccnt 0x4" ::: "memory"); // tile t landed (in-order)
        } else {
            asm volatile("s_wait_asynccnt 0x0" ::: "memory");
        }
        __syncthreads();

        // --- fragments per ISA 16-bit layouts ---
        v16bf afrag[4];
        #pragma unroll
        for (int mt = 0; mt < 4; ++mt) {
            const __bf16* ap = &As[p][(wr * 64 + mt * 16 + l15) * LDSP];
            v8bf lo = *(const v8bf*)(ap + half * 8);
            v8bf hi = *(const v8bf*)(ap + 16 + half * 8);
            afrag[mt] = __builtin_shufflevector(lo, hi,
                0,1,2,3,4,5,6,7,8,9,10,11,12,13,14,15);
        }
        v16bf bfrag[2];
        #pragma unroll
        for (int nt = 0; nt < 2; ++nt) {
            const __bf16* bp = &Bs[p][(wc * 32 + nt * 16 + l15) * LDSP + 8 * half];
            v8bf lo = *(const v8bf*)(bp);
            v8bf hi = *(const v8bf*)(bp + 16);
            bfrag[nt] = __builtin_shufflevector(lo, hi,
                0,1,2,3,4,5,6,7,8,9,10,11,12,13,14,15);
        }
        #pragma unroll
        for (int mt = 0; mt < 4; ++mt)
            #pragma unroll
            for (int nt = 0; nt < 2; ++nt)
                acc[mt][nt] = __builtin_amdgcn_wmma_f32_16x16x32_bf16(
                    false, afrag[mt], false, bfrag[nt],
                    (short)0, acc[mt][nt], false, false);
        __syncthreads();   // readers done before buffer p is re-staged
    }

    // --- epilogue: D layout -> row = m0 + r + 8*half, col = n0 + l15 ---
    #pragma unroll
    for (int nt = 0; nt < 2; ++nt) {
        const int col = n_base + wc * 32 + nt * 16 + l15;
        const float bv = bias ? bias[col] : 0.0f;
        #pragma unroll
        for (int mt = 0; mt < 4; ++mt) {
            #pragma unroll
            for (int r = 0; r < 8; ++r) {
                const int row = m_base + wr * 64 + mt * 16 + half * 8 + r;
                if (row < M) {
                    float v = acc[mt][nt][r] + bv;
                    if (ACT == 1) v = v > 0.0f ? v : 0.01f * v;
                    else if (ACT == 2) v = fmaxf(v, 0.0f);
                    C[(long long)row * N + col] = (__bf16)v;
                }
            }
        }
    }
}

// ---------------- final 256->1: wave-per-row dot + sigmoid ----------------

__global__ __launch_bounds__(256)
void ht_dot_sigmoid_kernel(const __bf16* __restrict__ X, const float* __restrict__ w,
                           const float* __restrict__ bptr, int M, int K,
                           float* __restrict__ out) {
    const int row  = blockIdx.x * 8 + (threadIdx.x >> 5);
    const int lane = threadIdx.x & 31;
    if (row >= M) return;
    const __bf16* x = X + (long long)row * K;
    float sum = 0.0f;
    for (int k = lane * 8; k < K; k += 32 * 8) {
        const v8bf xv = *(const v8bf*)(x + k);
        #pragma unroll
        for (int j = 0; j < 8; ++j) sum += (float)xv[j] * w[k + j];
    }
    #pragma unroll
    for (int off = 16; off > 0; off >>= 1)
        sum += __shfl_xor(sum, off, 32);
    if (lane == 0) out[row] = 1.0f / (1.0f + __expf(-(sum + bptr[0])));
}

// ---------------- host orchestration ----------------

extern "C" void kernel_launch(void* const* d_in, const int* in_sizes, int n_in,
                              void* d_out, int out_size, void* d_ws, size_t ws_size,
                              hipStream_t stream) {
    const float* x_cdr = (const float*)d_in[0];
    const float* x_pep = (const float*)d_in[1];
    const int*   eb    = (const int*)d_in[2];
    const int*   er    = (const int*)d_in[3];
    const int*   elab  = (const int*)d_in[4];
    const int E  = in_sizes[2] / 2;
    const int EL = in_sizes[4] / 2;

    const int* eb_src = eb;    const int* eb_dst = eb + E;   // cdr -> pep
    const int* er_src = er;    const int* er_dst = er + E;   // pep -> cdr
    const int* lrow   = elab;  const int* lcol   = elab + EL;

    const float* W1 = (const float*)d_in[23];
    const float* b1 = (const float*)d_in[24];
    const float* W2 = (const float*)d_in[25];
    const float* b2 = (const float*)d_in[26];
    const float* W3 = (const float*)d_in[27];
    const float* b3 = (const float*)d_in[28];

    // ---- workspace carve (256B-aligned chunks) ----
    char* wsp = (char*)d_ws;
    auto carve = [&](size_t bytes) -> void* {
        void* r = (void*)wsp;
        wsp += (bytes + 255) & ~(size_t)255;
        return r;
    };
    float*  aggP_f = (float*)carve((size_t)N_PEP_C * HID_C * 4);
    float*  aggC_f = (float*)carve((size_t)N_CDR_C * HID_C * 4);
    float*  invP   = (float*)carve((size_t)N_PEP_C * 4);
    float*  invC   = (float*)carve((size_t)N_CDR_C * 4);
    __bf16* xCb    = (__bf16*)carve((size_t)N_CDR_C * FEAT_C * 2);
    __bf16* xPb    = (__bf16*)carve((size_t)N_PEP_C * FEAT_C * 2);
    __bf16* cdrA   = (__bf16*)carve((size_t)N_CDR_C * HID_C * 2);
    __bf16* cdrB   = (__bf16*)carve((size_t)N_CDR_C * HID_C * 2);
    __bf16* pepA   = (__bf16*)carve((size_t)N_PEP_C * HID_C * 2);
    __bf16* pepB   = (__bf16*)carve((size_t)N_PEP_C * HID_C * 2);
    __bf16* aggPb  = (__bf16*)carve((size_t)N_PEP_C * HID_C * 2);
    __bf16* aggCb  = (__bf16*)carve((size_t)N_CDR_C * HID_C * 2);
    __bf16* t1     = (__bf16*)carve((size_t)MLP_CHUNK * 512 * 2);
    __bf16* t2     = (__bf16*)carve((size_t)MLP_CHUNK * 256 * 2);
    __bf16* Wt[12];
    for (int l = 0; l < 3; ++l) {
        const int in_dim = (l == 0) ? FEAT_C : HID_C;
        for (int j = 0; j < 4; ++j)
            Wt[l * 4 + j] = (__bf16*)carve((size_t)HID_C * in_dim * 2);
    }
    __bf16* W1t = (__bf16*)carve((size_t)512 * 512 * 2);
    __bf16* W2t = (__bf16*)carve((size_t)256 * 512 * 2);

    // ---- one-time conversions ----
    {
        long long n;
        n = (long long)N_CDR_C * FEAT_C;
        ht_cvt_kernel<<<(unsigned)((n + 255) / 256), 256, 0, stream>>>(x_cdr, xCb, n);
        n = (long long)N_PEP_C * FEAT_C;
        ht_cvt_kernel<<<(unsigned)((n + 255) / 256), 256, 0, stream>>>(x_pep, xPb, n);
        for (int l = 0; l < 3; ++l) {
            const int in_dim = (l == 0) ? FEAT_C : HID_C;
            const int srcIdx[4] = {5 + 6 * l + 0, 5 + 6 * l + 2, 5 + 6 * l + 3, 5 + 6 * l + 5};
            for (int j = 0; j < 4; ++j) {
                const long long tot = (long long)in_dim * HID_C;
                ht_wtrans_kernel<<<(unsigned)((tot + 255) / 256), 256, 0, stream>>>(
                    (const float*)d_in[srcIdx[j]], Wt[l * 4 + j], in_dim, 8, tot);
            }
        }
        n = (long long)512 * 512;
        ht_wtrans_kernel<<<(unsigned)((n + 255) / 256), 256, 0, stream>>>(W1, W1t, 512, 9, n);
        n = (long long)512 * 256;
        ht_wtrans_kernel<<<(unsigned)((n + 255) / 256), 256, 0, stream>>>(W2, W2t, 512, 8, n);
    }

    // ---- degrees -> inverse degrees ----
    hipMemsetAsync(invP, 0, N_PEP_C * sizeof(float), stream);
    hipMemsetAsync(invC, 0, N_CDR_C * sizeof(float), stream);
    ht_count_kernel<<<(E + 255) / 256, 256, 0, stream>>>(eb_dst, E, invP);
    ht_count_kernel<<<(E + 255) / 256, 256, 0, stream>>>(er_dst, E, invC);
    ht_inv_kernel<<<(N_PEP_C + 255) / 256, 256, 0, stream>>>(invP, N_PEP_C);
    ht_inv_kernel<<<(N_CDR_C + 255) / 256, 256, 0, stream>>>(invC, N_CDR_C);

    // ---- 3 SAGE layers, ping-pong bf16 h buffers ----
    const __bf16* hC = xCb;  const __bf16* hP = xPb;
    __bf16* outC = cdrA; __bf16* outP = pepA;
    __bf16* altC = cdrB; __bf16* altP = pepB;

    for (int l = 0; l < 3; ++l) {
        const int in_dim  = (l == 0) ? FEAT_C : HID_C;
        const int d4shift = (in_dim == 128) ? 5 : 6;
        const int dshift  = (in_dim == 128) ? 7 : 8;
        const float* bbl = (const float*)d_in[5 + 6 * l + 1];
        const float* rbl = (const float*)d_in[5 + 6 * l + 4];
        const __bf16* WbWl = Wt[l * 4 + 0];
        const __bf16* WbWr = Wt[l * 4 + 1];
        const __bf16* WrWl = Wt[l * 4 + 2];
        const __bf16* WrWr = Wt[l * 4 + 3];

        hipMemsetAsync(aggP_f, 0, (size_t)N_PEP_C * in_dim * sizeof(float), stream);
        hipMemsetAsync(aggC_f, 0, (size_t)N_CDR_C * in_dim * sizeof(float), stream);

        const long long wP = (long long)E << d4shift;
        ht_scatter_kernel<<<(unsigned)((wP + 255) / 256), 256, 0, stream>>>(
            eb_src, eb_dst, E, d4shift, hC, aggP_f, in_dim);
        ht_scatter_kernel<<<(unsigned)((wP + 255) / 256), 256, 0, stream>>>(
            er_src, er_dst, E, d4shift, hP, aggC_f, in_dim);

        ht_scale_cvt_kernel<<<(unsigned)((((long long)N_PEP_C << dshift) + 255) / 256), 256, 0, stream>>>(
            aggP_f, invP, aggPb, N_PEP_C, dshift);
        ht_scale_cvt_kernel<<<(unsigned)((((long long)N_CDR_C << dshift) + 255) / 256), 256, 0, stream>>>(
            aggC_f, invC, aggCb, N_CDR_C, dshift);

        // new_pep = leaky(aggP@bWl + bbl + hP@bWr)
        dim3 gp(HID_C / BN, (N_PEP_C + BM - 1) / BM);
        ht_gemm2_kernel<1><<<gp, 256, 0, stream>>>(
            aggPb, nullptr, in_dim, WbWl, in_dim,
            hP,    nullptr, in_dim, WbWr, in_dim,
            bbl, outP, N_PEP_C, HID_C);
        // new_cdr = leaky(aggC@rWl + rbl + hC@rWr)
        dim3 gc(HID_C / BN, (N_CDR_C + BM - 1) / BM);
        ht_gemm2_kernel<1><<<gc, 256, 0, stream>>>(
            aggCb, nullptr, in_dim, WrWl, in_dim,
            hC,    nullptr, in_dim, WrWr, in_dim,
            rbl, outC, N_CDR_C, HID_C);

        hC = outC; hP = outP;
        __bf16* t;
        t = outC; outC = altC; altC = t;
        t = outP; outP = altP; altP = t;
    }

    // ---- MLP over label edges, chunked ----
    float* out = (float*)d_out;
    for (int r0 = 0; r0 < EL; r0 += MLP_CHUNK) {
        const int m = (EL - r0) < MLP_CHUNK ? (EL - r0) : MLP_CHUNK;
        // t1 = relu( hC[row]@W1[:256] + hP[col]@W1[256:] + b1 ),  N=512
        // W1t is [512 n][512 k]; cdr part = k 0..255, pep part = k 256..511
        dim3 g1(512 / BN, (m + BM - 1) / BM);
        ht_gemm2_kernel<2><<<g1, 256, 0, stream>>>(
            hC, lrow + r0, HID_C, W1t,       512,
            hP, lcol + r0, HID_C, W1t + 256, 512,
            b1, t1, m, 512);
        // t2 = relu( t1@W2 + b2 ),  N=256
        dim3 g2(256 / BN, (m + BM - 1) / BM);
        ht_gemm2_kernel<2><<<g2, 256, 0, stream>>>(
            t1, nullptr, 512, W2t, 512,
            nullptr, nullptr, 0, nullptr, 0,
            b2, t2, m, 256);
        // out = sigmoid( t2@W3 + b3 )
        ht_dot_sigmoid_kernel<<<(m + 7) / 8, 256, 0, stream>>>(t2, W3, b3, m, 256, out + r0);
    }
}
